// Model_28037546508531
// MI455X (gfx1250) — matrix operand order repaired and checked by reading, not compiled, not run on hardware
//
#include <hip/hip_runtime.h>

typedef __attribute__((ext_vector_type(16))) _Float16 v16h;
typedef __attribute__((ext_vector_type(8)))  _Float16 v8h;
typedef __attribute__((ext_vector_type(8)))  float    v8f;

// Problem constants (match reference)
#define BB   32
#define TT   512
#define KK   4
#define DD   256
#define EE   100000
#define MM   2048
#define KDIM 512     // 2*D : GEMM K
#define NDIM 1024    // 4*D : GEMM N
#define NEGV (-10000000000.0f)

#define MTILE 64     // edges per workgroup (4 WMMA M-strips of 16)
#define NSTRIPS 4    // MTILE / 16

// ---------------------------------------------------------------------------
// Kernel 0: convert Wg, Wv (512x1024 f32, row-major) into transposed f16
// layout wt[n][k] so each WMMA B-fragment lane reads two runs of 8 contiguous halfs.
// ---------------------------------------------------------------------------
__global__ __launch_bounds__(256) void convert_w(const float* __restrict__ Wg,
                                                 const float* __restrict__ Wv,
                                                 _Float16* __restrict__ wt) {
  const int total = NDIM * KDIM;
  int i = blockIdx.x * blockDim.x + threadIdx.x;
  if (i >= 2 * total) return;
  const float* src = (i < total) ? Wg : Wv;
  _Float16* dst = wt + ((i < total) ? 0 : total);
  int j = (i < total) ? i : i - total;
  int n = j >> 9;          // / KDIM
  int k = j & (KDIM - 1);
  dst[(size_t)n * KDIM + k] = (_Float16)src[(size_t)k * NDIM + n];
}

// ---------------------------------------------------------------------------
// Kernel 1: fused edge gated-MLP.
//   lemmas[e] = sum_n silu((x_e @ Wg)[n]) * (x_e @ Wv)[n] * Wo[n]
// One workgroup (8 waves) handles 64 edges (4 M-strips); each wave owns 8 of
// the 64 N-tiles and multiplies all 4 M-strips against each B fragment
// (4x B-register reuse -> L2 traffic on Wg/Wv drops from 12.5 GB to ~3 GB).
// WMMA f32_16x16x32_f16, f32 accumulation, fused silu/Wo epilogue.
// ---------------------------------------------------------------------------
__global__ __launch_bounds__(256) void edge_mlp(
    const float* __restrict__ token,     // B*T*K*D f32
    const _Float16* __restrict__ wtg,    // [NDIM][KDIM] f16 (transposed Wg)
    const _Float16* __restrict__ wtv,    // [NDIM][KDIM] f16 (transposed Wv)
    const float* __restrict__ Wo,        // NDIM
    const int* __restrict__ edge,        // 2*E (scope rows, then goal rows)
    float* __restrict__ lemmas)          // E
{
  __shared__ _Float16 xs[MTILE][KDIM + 8];  // pad 8 halfs: A-frag bank spread
  __shared__ float lem[MTILE];

  const int tid = threadIdx.x;
  const int e0  = blockIdx.x * MTILE;

  if (tid < MTILE) lem[tid] = 0.0f;

  // ---- Gather + f32->f16 convert: 64 edges x 512 cols = 8192 float4 loads --
  #pragma unroll
  for (int it = 0; it < 32; ++it) {
    int f4  = tid + it * 256;      // 0..8191
    int e   = f4 >> 7;             // /128 : edge row in tile (0..63)
    int col = (f4 & 127) * 4;      // column (0..508)
    int eg  = e0 + e;
    int r = 0;
    if (eg < EE) r = (col < DD) ? edge[eg] : edge[EE + eg];  // scope | goal
    int cc  = (col < DD) ? col : col - DD;
    float4 v = *(const float4*)(token + (size_t)r * (KK * DD) + cc);
    xs[e][col + 0] = (_Float16)v.x;
    xs[e][col + 1] = (_Float16)v.y;
    xs[e][col + 2] = (_Float16)v.z;
    xs[e][col + 3] = (_Float16)v.w;
  }
  __syncthreads();

  const int lane    = tid & 31;
  const int wave    = tid >> 5;
  const int halfSel = lane >> 4;   // 0: lanes 0-15, 1: lanes 16-31
  const int l15     = lane & 15;

  // A 16x32 f16 layout: all lanes hold row M=l15; K chunks:
  //   lanes 0-15 : K in [0,8) u [16,24);  lanes 16-31: K in [8,16) u [24,32)
  const int klo = halfSel * 8;
  // B 32x16 layout: lane column N=l15; same K chunks as A
  const int kb  = halfSel * 8;

  float lemPart[NSTRIPS][8];
  #pragma unroll
  for (int s = 0; s < NSTRIPS; ++s)
    #pragma unroll
    for (int j = 0; j < 8; ++j) lemPart[s][j] = 0.0f;

  const v8f zeroAcc = {};

  for (int nt = wave; nt < NDIM / 16; nt += 8) {
    v8f accG[NSTRIPS], accV[NSTRIPS];
    #pragma unroll
    for (int s = 0; s < NSTRIPS; ++s) { accG[s] = zeroAcc; accV[s] = zeroAcc; }

    const int n = nt * 16 + l15;
    const _Float16* wgcol = wtg + (size_t)n * KDIM;
    const _Float16* wvcol = wtv + (size_t)n * KDIM;

    for (int kt = 0; kt < KDIM; kt += 32) {
      // B fragments: halfs [kb,kb+8) and [kb+16,kb+24) of transposed weight column
      v8h bg0 = *(const v8h*)&wgcol[kt + kb];
      v8h bg1 = *(const v8h*)&wgcol[kt + kb + 16];
      v16h bg = __builtin_shufflevector(bg0, bg1,
               0,1,2,3,4,5,6,7,8,9,10,11,12,13,14,15);
      v8h bv0 = *(const v8h*)&wvcol[kt + kb];
      v8h bv1 = *(const v8h*)&wvcol[kt + kb + 16];
      v16h bv = __builtin_shufflevector(bv0, bv1,
               0,1,2,3,4,5,6,7,8,9,10,11,12,13,14,15);

      // 4 M-strips share each B fragment: 8 WMMAs per K-step
      #pragma unroll
      for (int s = 0; s < NSTRIPS; ++s) {
        v8h alo = *(const v8h*)&xs[s * 16 + l15][kt + klo];
        v8h ahi = *(const v8h*)&xs[s * 16 + l15][kt + klo + 16];
        v16h a = __builtin_shufflevector(alo, ahi,
                 0,1,2,3,4,5,6,7,8,9,10,11,12,13,14,15);
        accG[s] = __builtin_amdgcn_wmma_f32_16x16x32_f16(
            false, a, false, bg, (short)0, accG[s], false, false);
        accV[s] = __builtin_amdgcn_wmma_f32_16x16x32_f16(
            false, a, false, bv, (short)0, accV[s], false, false);
      }
    }

    // Fused epilogue: silu(g)*v*Wo[n]; acc element j is row M=(halfSel*8+j)
    const float wo = Wo[n];
    #pragma unroll
    for (int s = 0; s < NSTRIPS; ++s)
      #pragma unroll
      for (int j = 0; j < 8; ++j) {
        float g = accG[s][j];
        float sg = g / (1.0f + __expf(-g));
        lemPart[s][j] += sg * accV[s][j] * wo;
      }
  }

  // Reduce over the 16 lanes of each half (same M rows, different N columns)
  #pragma unroll
  for (int msk = 1; msk < 16; msk <<= 1)
    #pragma unroll
    for (int s = 0; s < NSTRIPS; ++s)
      #pragma unroll
      for (int j = 0; j < 8; ++j)
        lemPart[s][j] += __shfl_xor(lemPart[s][j], msk, 32);

  if (l15 == 0) {
    #pragma unroll
    for (int s = 0; s < NSTRIPS; ++s)
      #pragma unroll
      for (int j = 0; j < 8; ++j)
        atomicAdd(&lem[s * 16 + halfSel * 8 + j], lemPart[s][j]); // ds_add_f32
  }
  __syncthreads();
  if (tid < MTILE && (e0 + tid) < EE) lemmas[e0 + tid] = lem[tid];
}

// ---------------------------------------------------------------------------
// Kernel 2: lm_preds[m,t] = mask[b,t] ? dot(mask_reprs[m], type_reprs[b,t]) : NEG
// Per-row batch pointer kills B-matrix sharing -> plain float4 dot kernel
// (0.5 GFLOP, L2-resident; not worth WMMA).
// ---------------------------------------------------------------------------
__global__ __launch_bounds__(256) void lm_kernel(
    const float* __restrict__ token,
    const int* __restrict__ lm_indices,      // M, into B*T*K rows of D
    const int* __restrict__ batch_pointers,  // M, into B
    const unsigned char* __restrict__ mask,  // B*T bool
    float* __restrict__ out)                 // M*T
{
  __shared__ float mrep[DD];
  const int m   = blockIdx.x;
  const int tid = threadIdx.x;
  const int r   = lm_indices[m];
  const int b   = batch_pointers[m];

  if (tid < DD / 4)
    ((float4*)mrep)[tid] = *(const float4*)(token + (size_t)r * DD + tid * 4);
  __syncthreads();

  for (int t = tid; t < TT; t += blockDim.x) {
    const float4* cand = (const float4*)(token + (size_t)(b * TT + t) * (KK * DD));
    float acc = 0.0f;
    #pragma unroll 8
    for (int dq = 0; dq < DD / 4; ++dq) {
      float4 c  = cand[dq];
      float4 mv = ((const float4*)mrep)[dq];
      acc += c.x * mv.x + c.y * mv.y + c.z * mv.z + c.w * mv.w;
    }
    out[(size_t)m * TT + t] = mask[b * TT + t] ? acc : NEGV;
  }
}

// ---------------------------------------------------------------------------
extern "C" void kernel_launch(void* const* d_in, const int* in_sizes, int n_in,
                              void* d_out, int out_size, void* d_ws, size_t ws_size,
                              hipStream_t stream) {
  const float* token = (const float*)d_in[0];              // (B,T,K,D) f32
  const float* Wg    = (const float*)d_in[1];              // (512,1024) f32
  const float* Wv    = (const float*)d_in[2];              // (512,1024) f32
  const float* Wo    = (const float*)d_in[3];              // (1024,1)   f32
  const int*   edge  = (const int*)d_in[4];                // (2,E)
  const int*   lmi   = (const int*)d_in[5];                // (M,)
  const int*   bptr  = (const int*)d_in[6];                // (M,)
  const unsigned char* mask = (const unsigned char*)d_in[7]; // (B,T) bool

  float* out = (float*)d_out;            // [E lemmas | M*T lm_preds]
  _Float16* wt = (_Float16*)d_ws;        // 2 * NDIM*KDIM f16 = 2 MB scratch

  // 1) one-shot weight transpose/convert to f16
  convert_w<<<(2 * NDIM * KDIM + 255) / 256, 256, 0, stream>>>(Wg, Wv, wt);

  // 2) fused edge gated-MLP via WMMA, 64 edges per block (tail-guarded)
  edge_mlp<<<(EE + MTILE - 1) / MTILE, 256, 0, stream>>>(
      token, wt, wt + (size_t)NDIM * KDIM, Wo, edge, out);

  // 3) lm_preds
  lm_kernel<<<MM, 256, 0, stream>>>(token, lmi, bptr, mask, out + EE);
}
